// TwoStreamVitLoss_51513837748763
// MI455X (gfx1250) — matrix operand order repaired and checked by reading, not compiled, not run on hardware
//
#include <hip/hip_runtime.h>
#include <hip/hip_bf16.h>
#include <math.h>

typedef __attribute__((ext_vector_type(16))) _Float16 v16h;
typedef __attribute__((ext_vector_type(8)))  _Float16 v8h;
typedef __attribute__((ext_vector_type(8)))  float    v8f;

#define B_      32
#define N_      196
#define C_      768
#define P_      1536
#define RPAD    208                    // rows per batch padded to 13*16
#define NRPAD   (B_*RPAD)              // 6656 padded rows
#define NROW    (B_*N_)                // 6272 valid rows
#define KCH     (C_/32)                // 24 k-chunks of 32
#define TEN_H   ((size_t)NRPAD*C_)     // halfs per swizzled tensor
#define MSE_N   ((size_t)B_*N_*P_)     // 9,633,792
#define INV_TAU 1.25f

struct Ptrs4 { const float* p[4]; };

// ---------------- zero accumulators ----------------
__global__ void k_zero(float* acc) {
    int i = threadIdx.x;
    if (i < 132) acc[i] = 0.0f;
}

// ---------------- fused 4x MSE reduction (bandwidth-bound) ----------------
__global__ void k_mse(const float* __restrict__ p0, const float* __restrict__ t0,
                      const float* __restrict__ p1, const float* __restrict__ t1,
                      const float* __restrict__ p2, const float* __restrict__ t2,
                      const float* __restrict__ p3, const float* __restrict__ t3,
                      float* acc) {
    int pair = blockIdx.y;
    const float* p; const float* t;
    switch (pair) {
        case 0:  p = p0; t = t0; break;
        case 1:  p = p1; t = t1; break;
        case 2:  p = p2; t = t2; break;
        default: p = p3; t = t3; break;
    }
    size_t n4 = MSE_N / 4;
    float s = 0.0f;
    for (size_t i = (size_t)blockIdx.x * blockDim.x + threadIdx.x; i < n4;
         i += (size_t)gridDim.x * blockDim.x) {
        float4 a = ((const float4*)p)[i];
        float4 b = ((const float4*)t)[i];
        float dx = a.x - b.x, dy = a.y - b.y, dz = a.z - b.z, dw = a.w - b.w;
        s += dx*dx + dy*dy + dz*dz + dw*dw;
    }
    for (int o = 16; o > 0; o >>= 1) s += __shfl_xor(s, o, 32);
    __shared__ float red[8];
    int lane = threadIdx.x & 31, w = threadIdx.x >> 5;
    if (lane == 0) red[w] = s;
    __syncthreads();
    if (threadIdx.x == 0) {
        float tot = 0.0f;
        for (int i = 0; i < 8; i++) tot += red[i];
        atomicAdd(&acc[pair], tot);
    }
}

// ---------------- per-row inverse L2 norms (wave per row) ----------------
__global__ void k_norm(Ptrs4 srcs, float* norms) {
    int lane = threadIdx.x & 31;
    int w = threadIdx.x >> 5;
    int rowG = blockIdx.x * 8 + w;
    if (rowG >= 4 * NROW) return;
    int t = rowG / NROW, r = rowG % NROW;
    const float* src = srcs.p[t] + (size_t)r * C_;
    float s = 0.0f;
    for (int j = lane; j < C_; j += 32) { float v = src[j]; s += v * v; }
    for (int o = 16; o > 0; o >>= 1) s += __shfl_xor(s, o, 32);
    if (lane == 0) norms[rowG] = 1.0f / fmaxf(sqrtf(s), 1e-12f);
}

// ---------------- normalize + f16 convert + WMMA fragment swizzle ----------------
// Per tensor, chunk c = ((tile*24 + kc)*32 + lane)*2 + hisel owns 8 contiguous
// halfs at hbuf + c*8, so the GEMM kernel's v16h loads are 32B contiguous per
// lane (fully coalesced, L2-resident).
// A-tensors (q): lane group g holds K = {g*8..g*8+7, 16+g*8..16+g*8+7} (ISA A layout).
// B-tensors (k): lane group g holds K = {g*8..g*8+7, 16+g*8..16+g*8+7} (same as A).
__global__ void k_swizzle(Ptrs4 srcs, const float* __restrict__ norms,
                          _Float16* __restrict__ hbuf) {
    const int CHUNKS = NRPAD * C_ / 8;               // 638,976 per tensor
    long long id = (long long)blockIdx.x * blockDim.x + threadIdx.x;
    if (id >= (long long)4 * CHUNKS) return;
    int t = (int)(id / CHUNKS);
    int c = (int)(id % CHUNKS);
    int tile  = c / 1536;                            // 24 kc * 32 lanes * 2 = 1536 chunks/tile
    int rem   = c % 1536;
    int kc    = rem >> 6;
    int lane  = (rem >> 1) & 31;
    int hisel = rem & 1;
    int rowPad = tile * 16 + (lane & 15);
    int g = lane >> 4;
    int kbase;
    if ((t & 1) == 0) kbase = kc * 32 + hisel * 16 + g * 8;  // A layout (q operand)
    else              kbase = kc * 32 + hisel * 16 + g * 8;  // B layout (k operand)
    v8h out;
    int batch = rowPad / RPAD, rIn = rowPad % RPAD;
    if (rIn < N_) {
        int srcRow = batch * N_ + rIn;
        const float* sp = srcs.p[t] + (size_t)srcRow * C_ + kbase;
        float inv = norms[t * NROW + srcRow];
        #pragma unroll
        for (int j = 0; j < 8; j++) out[j] = (_Float16)(sp[j] * inv);
    } else {
        #pragma unroll
        for (int j = 0; j < 8; j++) out[j] = (_Float16)0.0f;
    }
    *((v8h*)(hbuf + (size_t)t * TEN_H + (size_t)c * 8)) = out;
}

// ---------------- WMMA GEMM fused with exp/sum epilogue ----------------
// Wave tile 64x64 (4x4 accumulators, 128 VGPRs) for register-level fragment
// reuse: 8 b128-pair loads feed 16 WMMAs (~32 FLOP/B from L2).
// Workgroup: 8 waves as 2(M) x 4(N) -> WG tile 128 x 256.
__global__ void __launch_bounds__(256)
k_ctr(const _Float16* __restrict__ hbuf, float* acc) {
    int pair = blockIdx.y;
    const _Float16* qh = hbuf + (size_t)(2 * pair)     * TEN_H;
    const _Float16* kh = hbuf + (size_t)(2 * pair + 1) * TEN_H;
    float* sumAll = acc + 4  + pair * 32;
    float* sumPos = acc + 68 + pair * 32;

    const int WGN = NRPAD / 256;                     // 26
    int wgM = blockIdx.x / WGN;
    int wgN = blockIdx.x % WGN;
    int lane = threadIdx.x & 31;
    int w = threadIdx.x >> 5;
    int wm = w >> 2, wn = w & 3;
    int mt0 = wgM * 8 + wm * 4;                      // base M tile-of-16 (wave: 4 tiles)
    int nt0 = wgN * 16 + wn * 4;                     // base N tile-of-16 (wave: 4 tiles)

    v8f zero = {0.f, 0.f, 0.f, 0.f, 0.f, 0.f, 0.f, 0.f};
    v8f accv[4][4];
    #pragma unroll
    for (int i = 0; i < 4; i++)
        #pragma unroll
        for (int j = 0; j < 4; j++) accv[i][j] = zero;

    for (int kc = 0; kc < KCH; kc++) {
        v16h a[4], b[4];
        #pragma unroll
        for (int i = 0; i < 4; i++)
            a[i] = *((const v16h*)(qh + ((size_t)(mt0 + i) * KCH + kc) * 512 + lane * 16));
        #pragma unroll
        for (int j = 0; j < 4; j++)
            b[j] = *((const v16h*)(kh + ((size_t)(nt0 + j) * KCH + kc) * 512 + lane * 16));
        #pragma unroll
        for (int i = 0; i < 4; i++)
            #pragma unroll
            for (int j = 0; j < 4; j++)
                accv[i][j] = __builtin_amdgcn_wmma_f32_16x16x32_f16(
                    false, a[i], false, b[j], (short)0, accv[i][j], false, false);
    }

    // Branch-free epilogue: e = exp(sim/tau), masked and routed with cndmask-
    // style selects (no EXEC manipulation). A 128-row M tile spans <=2 batches.
    int b0 = (wgM * 128) / RPAD;
    float all0 = 0.f, all1 = 0.f, pos0 = 0.f, pos1 = 0.f;
    int nlo = lane & 15, g = lane >> 4;
    #pragma unroll
    for (int i = 0; i < 4; i++) {
        #pragma unroll
        for (int v = 0; v < 8; v++) {
            int Midx = (mt0 + i) * 16 + v + g * 8;   // C layout: M = v + 8*(lane>=16)
            int mb = Midx / RPAD, mo = Midx % RPAD;
            float mmask = (mo < N_) ? 1.0f : 0.0f;
            bool isb0 = (mb == b0);
            #pragma unroll
            for (int j = 0; j < 4; j++) {
                int Nidx = (nt0 + j) * 16 + nlo;
                int nb = Nidx / RPAD, no = Nidx % RPAD;
                float wgt = (no < N_) ? mmask : 0.0f;
                float e = __expf(accv[i][j][v] * INV_TAU) * wgt;
                float d = (nb == mb) ? e : 0.0f;
                float e0 = isb0 ? e : 0.0f;
                float d0 = isb0 ? d : 0.0f;
                all0 += e0; all1 += e - e0;
                pos0 += d0; pos1 += d - d0;
            }
        }
    }
    for (int o = 16; o > 0; o >>= 1) {
        all0 += __shfl_xor(all0, o, 32); all1 += __shfl_xor(all1, o, 32);
        pos0 += __shfl_xor(pos0, o, 32); pos1 += __shfl_xor(pos1, o, 32);
    }
    __shared__ float red[8][4];
    if (lane == 0) { red[w][0] = all0; red[w][1] = all1; red[w][2] = pos0; red[w][3] = pos1; }
    __syncthreads();
    if (threadIdx.x == 0) {
        float a0 = 0.f, a1 = 0.f, p0 = 0.f, p1 = 0.f;
        for (int i = 0; i < 8; i++) { a0 += red[i][0]; a1 += red[i][1]; p0 += red[i][2]; p1 += red[i][3]; }
        if (a0 != 0.f) atomicAdd(&sumAll[b0], a0);
        if (p0 != 0.f) atomicAdd(&sumPos[b0], p0);
        if (b0 + 1 < B_) {
            if (a1 != 0.f) atomicAdd(&sumAll[b0 + 1], a1);
            if (p1 != 0.f) atomicAdd(&sumPos[b0 + 1], p1);
        }
    }
}

// ---------------- finalize scalar loss ----------------
__global__ void k_final(const float* __restrict__ acc, float* out) {
    if (threadIdx.x == 0 && blockIdx.x == 0) {
        float loss = 0.0f;
        for (int i = 0; i < 4; i++) loss += acc[i] / (float)MSE_N;
        for (int p = 0; p < 2; p++) {
            float m = 0.0f;
            for (int b = 0; b < B_; b++) {
                float nce = logf(acc[4 + p * 32 + b]) - logf(acc[68 + p * 32 + b]);
                if (isnan(nce)) nce = 0.0f;
                m += nce;
            }
            loss += m / (float)B_;
        }
        out[0] = loss;
    }
}

extern "C" void kernel_launch(void* const* d_in, const int* in_sizes, int n_in,
                              void* d_out, int out_size, void* d_ws, size_t ws_size,
                              hipStream_t stream) {
    const float* rgb_rgb_hat   = (const float*)d_in[0];
    const float* rgb_flow_hat  = (const float*)d_in[1];
    const float* flow_rgb_hat  = (const float*)d_in[2];
    const float* flow_flow_hat = (const float*)d_in[3];
    const float* rgb_vis       = (const float*)d_in[4];
    const float* flow_vis      = (const float*)d_in[5];
    const float* rgb_token     = (const float*)d_in[6];
    const float* flow_token    = (const float*)d_in[7];
    const float* rgb_target    = (const float*)d_in[8];
    const float* flow_target   = (const float*)d_in[9];

    // Workspace: [0..255] f32 accumulators (4 MSE + 2x32 sumAll + 2x32 sumPos),
    // then 4*6272 f32 inv-norms, then 4 swizzled f16 tensors (~41 MB total).
    float* acc   = (float*)d_ws;
    float* norms = acc + 256;
    _Float16* hbuf = (_Float16*)(norms + 4 * NROW);

    k_zero<<<1, 256, 0, stream>>>(acc);

    dim3 mg(2048, 4);
    k_mse<<<mg, 256, 0, stream>>>(rgb_rgb_hat,  rgb_target,
                                  flow_flow_hat, flow_target,
                                  rgb_flow_hat,  flow_target,
                                  flow_rgb_hat,  rgb_target, acc);

    Ptrs4 s;
    s.p[0] = rgb_vis;  s.p[1] = flow_token;   // pair 0: q, k
    s.p[2] = flow_vis; s.p[3] = rgb_token;    // pair 1: q, k

    k_norm<<<(4 * NROW) / 8, 256, 0, stream>>>(s, norms);

    long long chunks = 4LL * NRPAD * C_ / 8;
    k_swizzle<<<(int)((chunks + 255) / 256), 256, 0, stream>>>(s, norms, hbuf);

    dim3 gg((NRPAD / 128) * (NRPAD / 256), 2);
    k_ctr<<<gg, 256, 0, stream>>>(hbuf, acc);

    k_final<<<1, 1, 0, stream>>>(acc, (float*)d_out);
}
